// CSWinAttention_85315230368229
// MI455X (gfx1250) — matrix operand order repaired and checked by reading, not compiled, not run on hardware
//
#include <hip/hip_runtime.h>
#include <hip/hip_bf16.h>

// ---------------------------------------------------------------------------
// CSWin attention forward for B=2, H=W=64, N=4, DIM=128, HEADS=4, SPLIT=2.
// Windows: hs=64, ws=2 -> 64 windows, L=512 tokens, head_dim=32.
// Kernel 1: fused flash attention (bf16 WMMA 16x16x32, f32 accum).
//   - row-max reduction via DPP ROW_XMASK (no ds_bpermute)
//   - row-sum via WMMA with an all-ones B matrix (l kept as a C-fragment)
// Kernel 2: depthwise-conv RPE term added into the output.
// ---------------------------------------------------------------------------

#define WINDOWS 64
#define NHEADS  4
#define L_TOK   512
#define HD      32
#define CHUNK   64
#define NCHUNK  (L_TOK / CHUNK)

typedef __attribute__((ext_vector_type(16))) __bf16 v16bf;
typedef __attribute__((ext_vector_type(8)))  __bf16 v8bf;
typedef __attribute__((ext_vector_type(8)))  float  v8f;

// Flat index of element (window w, token tok, channel = head*32+dim) in the
// (B, Ht, Wd, N, C) f32 tensors.  tok = (hp*2 + wp)*4 + nn.
__device__ __forceinline__ int gidx(int w, int head, int tok, int dim) {
  int b  = w >> 5;
  int j  = w & 31;
  int nn = tok & 3;
  int wp = (tok >> 2) & 1;
  int hp = tok >> 3;
  int wd = j * 2 + wp;
  return (((b * 64 + hp) * 64 + wd) * 4 + nn) * 128 + head * 32 + dim;
}

__device__ __forceinline__ v16bf ld16bf(const __bf16* p) {
  v8bf lo = *(const v8bf*)(p);
  v8bf hi = *(const v8bf*)(p + 16);
  return __builtin_shufflevector(lo, hi, 0,1,2,3,4,5,6,7,8,9,10,11,12,13,14,15);
}

__device__ __forceinline__ v16bf cvt16(float4 a, float4 b, float4 c, float4 d, float s) {
  v16bf r;
  r[0]  = (__bf16)(a.x * s); r[1]  = (__bf16)(a.y * s);
  r[2]  = (__bf16)(a.z * s); r[3]  = (__bf16)(a.w * s);
  r[4]  = (__bf16)(b.x * s); r[5]  = (__bf16)(b.y * s);
  r[6]  = (__bf16)(b.z * s); r[7]  = (__bf16)(b.w * s);
  r[8]  = (__bf16)(c.x * s); r[9]  = (__bf16)(c.y * s);
  r[10] = (__bf16)(c.z * s); r[11] = (__bf16)(c.w * s);
  r[12] = (__bf16)(d.x * s); r[13] = (__bf16)(d.y * s);
  r[14] = (__bf16)(d.z * s); r[15] = (__bf16)(d.w * s);
  return r;
}

__device__ __forceinline__ v8f wmma_bf16(v16bf a, v16bf b, v8f c) {
  return __builtin_amdgcn_wmma_f32_16x16x32_bf16(
      /*neg_a=*/false, a, /*neg_b=*/false, b,
      /*c_mod=*/(short)0, c, /*reuse_a=*/false, /*reuse_b=*/false);
}

// max with lane (lane ^ MASK) inside each 16-lane row via DPP ROW_XMASK.
template <int MASK>
__device__ __forceinline__ float dpp_xmask_max(float x) {
  int xi = __float_as_int(x);
  int t  = __builtin_amdgcn_update_dpp(xi, xi, 0x160 | MASK, 0xF, 0xF, false);
  return fmaxf(x, __int_as_float(t));
}

// full 16-lane-row max reduction (masks 1,2,4,8 stay inside each half-wave)
__device__ __forceinline__ float row16_max(float x) {
  x = dpp_xmask_max<1>(x);
  x = dpp_xmask_max<2>(x);
  x = dpp_xmask_max<4>(x);
  x = dpp_xmask_max<8>(x);
  return x;
}

// ---------------------------------------------------------------------------
// Flash attention: grid (4 row-chunks, 4 heads, 64 windows), 128 threads.
// Each wave owns two 16-row M-tiles; keys stream through LDS in 64-token
// chunks, double buffered.  LDS: K 8KB + Vt 8KB + P-stage 8KB = 24KB.
// ---------------------------------------------------------------------------
__global__ __launch_bounds__(128)
void cswin_attn_kernel(const float* __restrict__ Q, const float* __restrict__ K,
                       const float* __restrict__ V, float* __restrict__ Out) {
  __shared__ __bf16 Ks [2][CHUNK][HD];    // [buf][token][dim]
  __shared__ __bf16 Vts[2][HD][CHUNK];    // [buf][dim][token]  (transposed)
  __shared__ float  Pst[4][16][32];       // per-wave P re-layout tile

  const int rc   = blockIdx.x;            // row chunk: 128 query rows
  const int head = blockIdx.y;
  const int w    = blockIdx.z;
  const int tid  = threadIdx.x;
  const int lane = tid & 31;
  const int wv   = tid >> 5;
  const int half = lane >> 4;             // 0: lanes 0-15, 1: lanes 16-31
  const int l16  = lane & 15;

  const float SCALE = 0.17677669529663687f;   // 32^-0.5

  // all-ones B matrix: P x ones = row sums replicated over all 16 columns,
  // delivered directly in C-fragment layout (same as o / l).
  v16bf ones;
#pragma unroll
  for (int i = 0; i < 16; ++i) ones[i] = (__bf16)1.0f;

  // ---- per-wave persistent state: two M-tiles -----------------------------
  v16bf aq[2];
  float m_run[2][8];
  v8f   l_frag[2];
  v8f   o[2][2];

#pragma unroll
  for (int mi = 0; mi < 2; ++mi) {
    const int qtok = rc * 128 + (wv * 2 + mi) * 16 + l16;     // A row = lane%16
    const int kb   = half * 8;                                // A K-chunk base
    const float* qp = Q + gidx(w, head, qtok, 0);
    float4 q0 = *(const float4*)(qp + kb);
    float4 q1 = *(const float4*)(qp + kb + 4);
    float4 q2 = *(const float4*)(qp + kb + 16);
    float4 q3 = *(const float4*)(qp + kb + 20);
    aq[mi] = cvt16(q0, q1, q2, q3, SCALE);
#pragma unroll
    for (int r = 0; r < 8; ++r) m_run[mi][r] = -1e30f;
    l_frag[mi] = (v8f){};
    o[mi][0]   = (v8f){};
    o[mi][1]   = (v8f){};
  }

  // ---- chunk loader (global f32 -> regs -> bf16 LDS) ----------------------
  float kreg[16], vreg[16];
  const int dm = tid & 31;                 // channel within head (coalesced)
  auto issue_loads = [&](int c) {
#pragma unroll
    for (int i = 0; i < 16; ++i) {
      int tl = (tid >> 5) + i * 4;         // chunk-local token 0..63
      int g  = gidx(w, head, c * CHUNK + tl, dm);
      kreg[i] = K[g];
      vreg[i] = V[g];
    }
  };
  auto commit = [&](int buf) {
#pragma unroll
    for (int i = 0; i < 16; ++i) {
      int tl = (tid >> 5) + i * 4;
      Ks [buf][tl][dm] = (__bf16)kreg[i];
      Vts[buf][dm][tl] = (__bf16)vreg[i];
    }
  };

  issue_loads(0);
  commit(0);
  __syncthreads();

  // ---- main flash loop ----------------------------------------------------
#pragma unroll 2
  for (int c = 0; c < NCHUNK; ++c) {
    const int buf = c & 1;
    if (c + 1 < NCHUNK) issue_loads(c + 1);   // overlap with compute below

#pragma unroll
    for (int mi = 0; mi < 2; ++mi) {
      const int qtb = rc * 128 + (wv * 2 + mi) * 16;
#pragma unroll
      for (int kk = 0; kk < CHUNK; kk += 32) {
        // --- S = Q * K^T over 32 keys (two 16-col fragments) ---
        v16bf bk0 = ld16bf(&Ks[buf][kk      + l16][half * 8]);
        v16bf bk1 = ld16bf(&Ks[buf][kk + 16 + l16][half * 8]);
        v8f p0 = wmma_bf16(aq[mi], bk0, (v8f){});
        v8f p1 = wmma_bf16(aq[mi], bk1, (v8f){});

        const int kg0 = c * CHUNK + kk + l16;   // absolute key token (frag0)
        const int kg1 = kg0 + 16;               // absolute key token (frag1)

        // --- mask + online softmax (branch-free; EXEC stays all-ones) ---
#pragma unroll
        for (int r = 0; r < 8; ++r) {
          const int qt = qtb + half * 8 + r;    // absolute query token
          bool msk0 = ((qt >> 2) == (kg0 >> 2)) && (qt != kg0);
          bool msk1 = ((qt >> 2) == (kg1 >> 2)) && (qt != kg1);
          p0[r] = msk0 ? -1e30f : p0[r];
          p1[r] = msk1 ? -1e30f : p1[r];

          float tm   = row16_max(fmaxf(p0[r], p1[r]));
          float mnew = fmaxf(m_run[mi][r], tm);
          float corr = __expf(m_run[mi][r] - mnew);
          p0[r] = __expf(p0[r] - mnew);
          p1[r] = __expf(p1[r] - mnew);
          m_run[mi][r]   = mnew;
          l_frag[mi][r] *= corr;
          o[mi][0][r]   *= corr;
          o[mi][1][r]   *= corr;
        }

        // --- re-layout P: C-frag -> A-frag via per-wave LDS tile ---
#pragma unroll
        for (int r = 0; r < 8; ++r) {
          Pst[wv][half * 8 + r][l16]      = p0[r];
          Pst[wv][half * 8 + r][16 + l16] = p1[r];
        }
        const float* pr = &Pst[wv][l16][half * 8];
        float4 pa = *(const float4*)(pr);
        float4 pb = *(const float4*)(pr + 4);
        float4 pc = *(const float4*)(pr + 16);
        float4 pd = *(const float4*)(pr + 20);
        v16bf ap = cvt16(pa, pb, pc, pd, 1.0f);

        // --- row sums: l += P * ones (replicated across columns) ---
        l_frag[mi] = wmma_bf16(ap, ones, l_frag[mi]);

        // --- O += P * V (two 16-dim output fragments) ---
        v16bf bv0 = ld16bf(&Vts[buf][l16]     [kk + half * 8]);
        v16bf bv1 = ld16bf(&Vts[buf][16 + l16][kk + half * 8]);
        o[mi][0] = wmma_bf16(ap, bv0, o[mi][0]);
        o[mi][1] = wmma_bf16(ap, bv1, o[mi][1]);
      }
    }

    if (c + 1 < NCHUNK) commit((c + 1) & 1);
    __syncthreads();
  }

  // ---- normalize + scatter to (B,Ht,Wd,N,C) output ------------------------
#pragma unroll
  for (int mi = 0; mi < 2; ++mi) {
    const int qtb = rc * 128 + (wv * 2 + mi) * 16;
#pragma unroll
    for (int r = 0; r < 8; ++r) {
      const int qt  = qtb + half * 8 + r;
      const float inv = 1.0f / l_frag[mi][r];
      const int base = gidx(w, head, qt, 0);
      Out[base + l16]      = o[mi][0][r] * inv;
      Out[base + 16 + l16] = o[mi][1][r] * inv;
    }
  }
}

// ---------------------------------------------------------------------------
// RPE: depthwise 3x3 conv over each (64 x 2) window image, summed over n,
// plus center-tap cross-n term; accumulated into Out (runs after attention).
// One thread per (window, hp, wp, c); channels fastest for coalescing.
// ---------------------------------------------------------------------------
__global__ __launch_bounds__(256)
void cswin_rpe_kernel(const float* __restrict__ V, const float* __restrict__ Wc,
                      float* __restrict__ Out) {
  int g  = blockIdx.x * 256 + threadIdx.x;   // 64*64*2*128 = 1048576 threads
  int c  = g & 127;
  int wp = (g >> 7) & 1;
  int hp = (g >> 8) & 63;
  int w  = g >> 14;
  int b  = w >> 5;
  int j  = w & 31;

  float w9[9];
#pragma unroll
  for (int t = 0; t < 9; ++t) w9[t] = Wc[c * 9 + t];
  const float center = w9[4];

  float conv_sum = 0.0f;
  float vc[4];
  float vcsum = 0.0f;
#pragma unroll
  for (int nn = 0; nn < 4; ++nn) {
    float acc = 0.0f;
#pragma unroll
    for (int dy = -1; dy <= 1; ++dy) {
      int y = hp + dy;
      if (y < 0 || y >= 64) continue;
#pragma unroll
      for (int dx = -1; dx <= 1; ++dx) {
        int x = wp + dx;
        if (x < 0 || x >= 2) continue;
        float v = V[(((b * 64 + y) * 64 + (j * 2 + x)) * 4 + nn) * 128 + c];
        acc += w9[(dy + 1) * 3 + (dx + 1)] * v;
      }
    }
    conv_sum += acc;
    float vcn = V[(((b * 64 + hp) * 64 + (j * 2 + wp)) * 4 + nn) * 128 + c];
    vc[nn] = vcn;
    vcsum += vcn;
  }

  const int obase = (((b * 64 + hp) * 64 + (j * 2 + wp)) * 4) * 128 + c;
#pragma unroll
  for (int nn = 0; nn < 4; ++nn) {
    Out[obase + nn * 128] += conv_sum + center * (vc[nn] - vcsum);
  }
}

// ---------------------------------------------------------------------------
extern "C" void kernel_launch(void* const* d_in, const int* in_sizes, int n_in,
                              void* d_out, int out_size, void* d_ws, size_t ws_size,
                              hipStream_t stream) {
  (void)in_sizes; (void)n_in; (void)out_size; (void)d_ws; (void)ws_size;
  const float* Q  = (const float*)d_in[0];
  const float* K  = (const float*)d_in[1];
  const float* V  = (const float*)d_in[2];
  const float* Wc = (const float*)d_in[3];
  float* Out = (float*)d_out;

  dim3 grid(4, NHEADS, WINDOWS);           // row-chunk, head, window
  cswin_attn_kernel<<<grid, 128, 0, stream>>>(Q, K, V, Out);

  const int rpe_threads = WINDOWS * 64 * 2 * 128;   // 1,048,576
  cswin_rpe_kernel<<<rpe_threads / 256, 256, 0, stream>>>(V, Wc, Out);
}
